// head_86603720557255
// MI455X (gfx1250) — matrix operand order repaired and checked by reading, not compiled, not run on hardware
//
#include <hip/hip_runtime.h>
#include <hip/hip_bf16.h>

// CDNA5 / gfx1250: wave32, WMMA 16x16x32 bf16 -> f32
typedef __attribute__((ext_vector_type(16))) __bf16 v16bf;
typedef __attribute__((ext_vector_type(8)))  __bf16 v8bf;   // 16 bytes
typedef __attribute__((ext_vector_type(2)))  __bf16 v2bf;   // 4 bytes
typedef __attribute__((ext_vector_type(8)))  float  v8f;

#define B_  16
#define S_  2048
#define DM_ 1024
#define DH_ 128

union fragU { v16bf v; v8bf h[2]; };

// Build a 16-element bf16 fragment from two contiguous 8-element (16B) runs.
__device__ __forceinline__ v16bf load_frag(const __bf16* p0, const __bf16* p1) {
  fragU u;
  u.h[0] = *(const v8bf*)p0;
  u.h[1] = *(const v8bf*)p1;
  return u.v;
}

// A-matrix (16-bit, 16x32) per-lane K runs: [8*half, +7] and [16+8*half, +7]
__device__ __forceinline__ v16bf load_frag_A(const __bf16* rowbase, int half) {
  return load_frag(rowbase + 8 * half, rowbase + 16 + 8 * half);
}
// B-matrix (16-bit, 32x16) per-lane K runs: [8*half, +7] and [16+8*half, +7]
__device__ __forceinline__ v16bf load_frag_B(const __bf16* colbase, int half) {
  return load_frag(colbase + 8 * half, colbase + 16 + 8 * half);
}

__device__ __forceinline__ v8f wmma_bf16(v16bf a, v16bf b, v8f c) {
  return __builtin_amdgcn_wmma_f32_16x16x32_bf16(
      false, a, false, b, (short)0, c, false, false);
}

// ---------------------------------------------------------------------------
// Kernel 1: qkv projection.  out = (x @ W + b) * scale, stored bf16.
// grid = (M/128, 3), block = 256 (8 waves). blockIdx.y: 0=q, 1=k, 2=v.
// ---------------------------------------------------------------------------
__global__ __launch_bounds__(256) void qkv_proj_kernel(
    const float* __restrict__ x,
    const float* __restrict__ Wq, const float* __restrict__ bq,
    const float* __restrict__ Wk, const float* __restrict__ bk,
    const float* __restrict__ Wv, const float* __restrict__ bv,
    __bf16* __restrict__ qb, __bf16* __restrict__ kb, __bf16* __restrict__ vb)
{
  const float* W;  const float* bias;  __bf16* out;  float scale;
  if (blockIdx.y == 0)      { W = Wq; bias = bq; out = qb; scale = 0.03125f; } // 1/sqrt(1024)
  else if (blockIdx.y == 1) { W = Wk; bias = bk; out = kb; scale = 1.0f; }
  else                      { W = Wv; bias = bv; out = vb; scale = 1.0f; }

  // Rows padded to 40 bf16 (80 B, 16B multiple) so v8bf LDS loads are aligned.
  __shared__ __align__(16) __bf16 xs[128][40];     // x tile: 128 rows x 32 K
  __shared__ __align__(16) __bf16 wls_t[128][40];  // W tile TRANSPOSED: [n][k]

  const int tid  = threadIdx.x;
  const int wave = tid >> 5, lane = tid & 31;
  const int half = lane >> 4, l16 = lane & 15;
  const int rg = wave >> 1, cg = wave & 1;         // 4 row-groups x 2 col-groups
  const int m0 = blockIdx.x * 128;

  const v8f zv = {0.f, 0.f, 0.f, 0.f, 0.f, 0.f, 0.f, 0.f};
  v8f acc[2][4];
  #pragma unroll
  for (int rt = 0; rt < 2; ++rt)
    #pragma unroll
    for (int ct = 0; ct < 4; ++ct) acc[rt][ct] = zv;

  for (int kt = 0; kt < DM_ / 32; ++kt) {
    // x chunk 128x32: f32-pair loads (b64), packed bf16-pair LDS stores (b32)
    #pragma unroll
    for (int i = 0; i < 8; ++i) {
      int p = tid + i * 256;                 // 0..2047 pairs, 16 pairs/row
      int r = p >> 4;
      int c = (p & 15) * 2;
      const float2 xv = *(const float2*)&x[(size_t)(m0 + r) * DM_ + (size_t)kt * 32 + c];
      v2bf pk = {(__bf16)xv.x, (__bf16)xv.y};
      *(v2bf*)&xs[r][c] = pk;
    }
    // W chunk 32x128 -> transposed LDS wls_t[n][k], packed pairs along k
    #pragma unroll
    for (int i = 0; i < 8; ++i) {
      int p = tid + i * 256;                 // 0..2047
      int n = p & 127;                       // coalesced over n
      int k = (p >> 7) * 2;                  // 0,2,..,30
      float w0 = W[(size_t)(kt * 32 + k) * DH_ + n];
      float w1 = W[(size_t)(kt * 32 + k + 1) * DH_ + n];
      v2bf pk = {(__bf16)w0, (__bf16)w1};
      *(v2bf*)&wls_t[n][k] = pk;
    }
    __syncthreads();

    v16bf af[2], bfm[4];
    #pragma unroll
    for (int rt = 0; rt < 2; ++rt)
      af[rt] = load_frag_A(&xs[rg * 32 + rt * 16 + l16][0], half);
    #pragma unroll
    for (int ct = 0; ct < 4; ++ct)
      bfm[ct] = load_frag_B(&wls_t[cg * 64 + ct * 16 + l16][0], half);

    #pragma unroll
    for (int rt = 0; rt < 2; ++rt)
      #pragma unroll
      for (int ct = 0; ct < 4; ++ct)
        acc[rt][ct] = wmma_bf16(af[rt], bfm[ct], acc[rt][ct]);
    __syncthreads();
  }

  // epilogue: + bias, * scale, store bf16 (C layout: M = v + 8*half, N = l16)
  #pragma unroll
  for (int rt = 0; rt < 2; ++rt) {
    #pragma unroll
    for (int ct = 0; ct < 4; ++ct) {
      int n = cg * 64 + ct * 16 + l16;
      float bn = bias[n];
      #pragma unroll
      for (int v = 0; v < 8; ++v) {
        int row = m0 + rg * 32 + rt * 16 + 8 * half + v;
        out[(size_t)row * DH_ + n] = (__bf16)((acc[rt][ct][v] + bn) * scale);
      }
    }
  }
}

// ---------------------------------------------------------------------------
// Kernel 2: causal flash attention.  q pre-scaled by 1/sqrt(d_model).
// grid = (S/128, B), block = 256 (8 waves); wave w owns 16 query rows.
// ---------------------------------------------------------------------------
__global__ __launch_bounds__(256) void fa_kernel(
    const __bf16* __restrict__ qb, const __bf16* __restrict__ kb,
    const __bf16* __restrict__ vb, float* __restrict__ out)
{
  __shared__ __align__(16) __bf16 ks[32][136];       // K chunk: [key][dh], 272B rows
  __shared__ __align__(16) __bf16 vst[128][40];      // V chunk TRANSPOSED: [dh][key]
  __shared__ __align__(16) __bf16 ps[8][16][40];     // per-wave P tile: [row][key]

  const int tid  = threadIdx.x;
  const int wave = tid >> 5, lane = tid & 31;
  const int half = lane >> 4, l16 = lane & 15;
  const int qbase = blockIdx.x * 128;
  const int b = blockIdx.y;
  const size_t base = (size_t)b * S_ * DH_;

  // q fragments: 4 K-steps of 32 over DH=128, A layout, 2x global b128 each
  v16bf qf[4];
  {
    const int qrow = qbase + wave * 16 + l16;
    const __bf16* qp = qb + base + (size_t)qrow * DH_;
    #pragma unroll
    for (int s = 0; s < 4; ++s)
      qf[s] = load_frag_A(qp + s * 32, half);
  }

  const v8f zv = {0.f, 0.f, 0.f, 0.f, 0.f, 0.f, 0.f, 0.f};
  float mI[8], lI[8];
  v8f oacc[8];
  #pragma unroll
  for (int v = 0; v < 8; ++v) { mI[v] = -3.0e38f; lI[v] = 0.f; }
  #pragma unroll
  for (int t = 0; t < 8; ++t) oacc[t] = zv;

  const int nch = qbase / 32 + 4;   // causal: keys up to qbase+127
  for (int j = 0; j < nch; ++j) {
    const int kbase = j * 32;

    // prefetch next chunk into cache while we work on this one
    if (j + 1 < nch) {
      size_t pg = base + (size_t)(kbase + 32) * DH_ + (size_t)(tid << 4 & 4095);
      __builtin_prefetch(kb + pg, 0, 0);
      __builtin_prefetch(vb + pg, 0, 0);
    }

    // cooperative load: b128 moves; K row-major, V transposed scatter
    #pragma unroll
    for (int i = 0; i < 2; ++i) {
      int o8 = tid + i * 256;               // 0..511 octets, 16 octets/row
      int r = o8 >> 4;
      int o = (o8 & 15) * 8;                // dh start
      size_t g = base + (size_t)(kbase + r) * DH_ + o;
      const v8bf kv = *(const v8bf*)&kb[g];
      *(v8bf*)&ks[r][o] = kv;
      const v8bf vv = *(const v8bf*)&vb[g];
      #pragma unroll
      for (int e = 0; e < 8; ++e) vst[o + e][r] = vv[e];
    }
    __syncthreads();

    // ---- gemm1: S (16x32 per wave) = q (16x128) @ K^T (128x32) ----
    v8f sacc[2] = {zv, zv};
    #pragma unroll
    for (int s = 0; s < 4; ++s) {
      #pragma unroll
      for (int ct = 0; ct < 2; ++ct) {
        // B[kdim][n] = K[key = ct*16+l16][dh = s*32 + 8*half + e (+8 if e>=8)]
        v16bf bfr = load_frag_B(&ks[ct * 16 + l16][s * 32], half);
        sacc[ct] = wmma_bf16(qf[s], bfr, sacc[ct]);
      }
    }

    // causal mask only on diagonal-touching chunks
    if (kbase + 31 > qbase + wave * 16) {
      #pragma unroll
      for (int ct = 0; ct < 2; ++ct)
        #pragma unroll
        for (int v = 0; v < 8; ++v) {
          int rowg = qbase + wave * 16 + 8 * half + v;
          int colg = kbase + ct * 16 + l16;
          if (colg > rowg) sacc[ct][v] = -3.0e38f;
        }
    }

    // ---- online softmax (row = reduce across 16-lane half-group) ----
    float p0[8], p1[8], cf[8];
    #pragma unroll
    for (int v = 0; v < 8; ++v) {
      float rm = fmaxf(sacc[0][v], sacc[1][v]);
      #pragma unroll
      for (int msk = 1; msk < 16; msk <<= 1)
        rm = fmaxf(rm, __shfl_xor(rm, msk, 32));
      float mn = fmaxf(mI[v], rm);
      float c  = __expf(mI[v] - mn);
      float e0 = __expf(sacc[0][v] - mn);
      float e1 = __expf(sacc[1][v] - mn);
      float rs = e0 + e1;
      #pragma unroll
      for (int msk = 1; msk < 16; msk <<= 1)
        rs += __shfl_xor(rs, msk, 32);
      lI[v] = lI[v] * c + rs;
      mI[v] = mn;
      cf[v] = c; p0[v] = e0; p1[v] = e1;
    }
    #pragma unroll
    for (int t = 0; t < 8; ++t)
      #pragma unroll
      for (int v = 0; v < 8; ++v) oacc[t][v] *= cf[v];

    // P: C layout -> LDS (row-major [row][key]), one bf16 per store
    #pragma unroll
    for (int v = 0; v < 8; ++v) {
      ps[wave][8 * half + v][l16]      = (__bf16)p0[v];
      ps[wave][8 * half + v][16 + l16] = (__bf16)p1[v];
    }
    __syncthreads();

    // ---- gemm2: O (16x128) += P (16x32) @ V (32x128) ----
    v16bf pf = load_frag_A(&ps[wave][l16][0], half);
    #pragma unroll
    for (int t = 0; t < 8; ++t) {
      // B[key][dh=n]: vst[dh = t*16+l16][key = 8*half + e (+8 if e>=8)]
      v16bf bfr = load_frag_B(&vst[t * 16 + l16][0], half);
      oacc[t] = wmma_bf16(pf, bfr, oacc[t]);
    }
    __syncthreads();   // before next chunk overwrites ks/vst
  }

  // epilogue: O / l  -> fp32 out
  #pragma unroll
  for (int v = 0; v < 8; ++v) {
    int rowg = qbase + wave * 16 + 8 * half + v;
    float inv = 1.0f / lI[v];
    float* op = out + base + (size_t)rowg * DH_;
    #pragma unroll
    for (int t = 0; t < 8; ++t) op[t * 16 + l16] = oacc[t][v] * inv;
  }
}

// ---------------------------------------------------------------------------
extern "C" void kernel_launch(void* const* d_in, const int* in_sizes, int n_in,
                              void* d_out, int out_size, void* d_ws, size_t ws_size,
                              hipStream_t stream) {
  const float* x  = (const float*)d_in[0];
  const float* Wq = (const float*)d_in[1];
  const float* bq = (const float*)d_in[2];
  const float* Wk = (const float*)d_in[3];
  const float* bk = (const float*)d_in[4];
  const float* Wv = (const float*)d_in[5];
  const float* bv = (const float*)d_in[6];

  __bf16* qb = (__bf16*)d_ws;                         // 3 x 8 MB bf16 scratch
  __bf16* kb = qb + (size_t)B_ * S_ * DH_;
  __bf16* vb = kb + (size_t)B_ * S_ * DH_;

  dim3 g1((B_ * S_) / 128, 3);
  qkv_proj_kernel<<<g1, 256, 0, stream>>>(x, Wq, bq, Wk, bk, Wv, bv, qb, kb, vb);

  dim3 g2(S_ / 128, B_);
  fa_kernel<<<g2, 256, 0, stream>>>(qb, kb, vb, (float*)d_out);
}
